// AttentiveAtlasEncoder_64914135711987
// MI455X (gfx1250) — compile-verified, matrix operand order checked by reading, not run on hardware
//
#include <hip/hip_runtime.h>
#include <cstdint>
#include <cstddef>

typedef __bf16 bf16;
typedef __attribute__((ext_vector_type(16))) __bf16 v16bf;
typedef __attribute__((ext_vector_type(8)))  float   v8f;

#define DEV static __device__ __forceinline__

// ---------------- constants / output layout (flat f32 elements) ----------------
static constexpr size_t NB = 32768;           // batch
static constexpr size_t OFF_KCHART    = 0;                       // (B,) int
static constexpr size_t OFF_KCODE     = NB;                      // (B,) int
static constexpr size_t OFF_ZN        = 2 * NB;                  // (B,16)
static constexpr size_t OFF_ZTEX      = OFF_ZN + 16 * NB;        // (B,16)
static constexpr size_t OFF_RW        = OFF_ZTEX + 16 * NB;      // (B,8)
static constexpr size_t OFF_ZGEO      = OFF_RW + 8 * NB;         // (B,16)
static constexpr size_t OFF_LOSS      = OFF_ZGEO + 16 * NB;      // scalar
static constexpr size_t OFF_IDX       = OFF_LOSS + 1;            // (B,8) int
static constexpr size_t OFF_ZNALL_OUT = OFF_IDX + 8 * NB;        // (B,8,16)
static constexpr size_t OFF_CBAR      = OFF_ZNALL_OUT + 128 * NB;// (B,16)

// workspace byte offsets
static constexpr size_t WS_W1C   = 0;        // bf16 [256][64]   col-major (N-major)
static constexpr size_t WS_W2C   = 32768;    // bf16 [256][256]  col-major
static constexpr size_t WS_WVC   = 163840;   // bf16 [16][256]   col-major
static constexpr size_t WS_CODEB = 172032;   // bf16 [512][32]   row-major, K padded to 32
static constexpr size_t WS_CNRM  = 204800;   // f32  [512]
static constexpr size_t WS_LOSSP = 206848;   // f32  [512] per-block loss partials

// ---------------- helpers ----------------
DEV float gelu_exact(float x) {
    return 0.5f * x * (1.0f + erff(x * 0.70710678118654752f));
}

// Load a 16-bit WMMA fragment (A from row-major [M][K], or B from N-major [N][K]).
// Per ISA 7.12.2: lane l<16 -> row l, K chunks [0..7],[16..23]; lane>=16 -> row l-16,
// chunks [8..15],[24..31]. Each chunk is 8 bf16 = one 16-byte load.
DEV v16bf ld_frag(const bf16* p, int strideElems, int lrow, int hi) {
    const bf16* r = p + (size_t)lrow * strideElems + hi * 8;
    union { v16bf v; uint4 q[2]; } u;
    u.q[0] = *reinterpret_cast<const uint4*>(r);
    u.q[1] = *reinterpret_cast<const uint4*>(r + 16);
    return u.v;
}

// Same fragment pattern but converting f32 source (for the x activations).
DEV v16bf ld_frag_f32(const float* p, int strideElems, int lrow, int hi) {
    const float* r = p + (size_t)lrow * strideElems + hi * 8;
    union { v16bf v; bf16 e[16]; } u;
#pragma unroll
    for (int i = 0; i < 8; ++i) { u.e[i] = (bf16)r[i]; u.e[8 + i] = (bf16)r[16 + i]; }
    return u.v;
}

DEV v8f wmma_bf16(v16bf a, v16bf b, v8f c) {
    return __builtin_amdgcn_wmma_f32_16x16x32_bf16(false, a, false, b, (short)0, c, false, false);
}

// ---------------- prep: weight conversion to bf16 WMMA-friendly layouts ----------------
__global__ void atlas_prep(const float* __restrict__ W1, const float* __restrict__ W2,
                           const float* __restrict__ Wv, const float* __restrict__ cbk,
                           bf16* __restrict__ W1c, bf16* __restrict__ W2c,
                           bf16* __restrict__ Wvc, bf16* __restrict__ codeB,
                           float* __restrict__ codeNrm) {
    int i = blockIdx.x * blockDim.x + threadIdx.x;  // 0..102911
    if (i < 16384) {                     // W1 [64][256] -> W1c[n*64+k]
        int n = i >> 6, k = i & 63;
        W1c[i] = (bf16)W1[k * 256 + n];
        return;
    }
    i -= 16384;
    if (i < 65536) {                     // W2 [256][256] -> W2c[n*256+k]
        int n = i >> 8, k = i & 255;
        W2c[i] = (bf16)W2[k * 256 + n];
        return;
    }
    i -= 65536;
    if (i < 4096) {                      // Wv [256][16] -> Wvc[n*256+k]
        int n = i >> 8, k = i & 255;
        Wvc[i] = (bf16)Wv[k * 16 + n];
        return;
    }
    i -= 4096;
    if (i < 16384) {                     // codebook [512][16] -> [512][32] zero-padded
        int code = i >> 5, k = i & 31;
        codeB[i] = (k < 16) ? (bf16)cbk[code * 16 + k] : (bf16)0.0f;
        return;
    }
    i -= 16384;
    if (i < 512) {                       // ||code||^2
        float s = 0.f;
#pragma unroll
        for (int d = 0; d < 16; ++d) { float v = cbk[i * 16 + d]; s += v * v; }
        codeNrm[i] = s;
    }
}

// ---------------- main fused kernel: 64 rows per block, 256 threads (8 waves) ----------------
__global__ __launch_bounds__(256)
void atlas_main(const float* __restrict__ x,
                const float* __restrict__ b1g, const float* __restrict__ b2g,
                const float* __restrict__ bvg, const float* __restrict__ ccg,
                const float* __restrict__ cbk,
                const float* __restrict__ sfW1, const float* __restrict__ sfb1,
                const float* __restrict__ sfW2, const float* __restrict__ sfb2,
                const bf16* __restrict__ W1c, const bf16* __restrict__ W2c,
                const bf16* __restrict__ Wvc, const bf16* __restrict__ codeB,
                const float* __restrict__ codeNrm,
                float* __restrict__ out, float* __restrict__ lossPart) {
    __shared__ __align__(16) char smem[87296];
    bf16*  sF1  = (bf16*)(smem + 0);       // [64][256] bf16 (GEMM1 out / GEMM2 A)
    bf16*  sF2  = (bf16*)(smem + 32768);   // [64][256] bf16 (GEMM2 out / GEMM3 A)
    float* sV   = (float*)(smem + 65536);  // [64][16]
    float* sRw  = (float*)(smem + 69632);  // [64][8]
    float* sCb  = (float*)(smem + 71680);  // [64][16]
    float* sVl  = (float*)(smem + 75776);  // [64][16] v_local f32
    bf16*  sVlB = (bf16*)(smem + 79872);   // [64][32] v_local bf16, K-padded (B of dist GEMM)
    int*   sIdx = (int*)(smem + 83968);    // [8][64]
    int*   sKch = (int*)(smem + 86016);    // [64]
    float* sRed = (float*)(smem + 86272);  // [256]
    float* sZnAll = (float*)(smem + 0);    // [64][8][16] f32 — aliases sF1 (dead by then)

    const int tid  = threadIdx.x;
    const int wv   = tid >> 5;
    const int lane = tid & 31;
    const int lrow = lane & 15;
    const int hi   = lane >> 4;
    const int rowBase = blockIdx.x * 64;

    __builtin_prefetch(W2c + (size_t)(wv * 32) * 256, 0, 3);

    // ================= GEMM1: f1 = gelu(x @ W1 + b1)  (K=64) =================
    {
        const int n0 = wv * 32;
        v16bf Bf[2][2];
#pragma unroll
        for (int ntl = 0; ntl < 2; ++ntl)
#pragma unroll
            for (int ks = 0; ks < 2; ++ks)
                Bf[ntl][ks] = ld_frag(W1c + (size_t)(n0 + ntl * 16) * 64 + ks * 32, 64, lrow, hi);
        float bias[2] = { b1g[n0 + lrow], b1g[n0 + 16 + lrow] };
#pragma unroll
        for (int mt = 0; mt < 4; ++mt) {
            v16bf Af[2];
#pragma unroll
            for (int ks = 0; ks < 2; ++ks)
                Af[ks] = ld_frag_f32(x + (size_t)(rowBase + mt * 16) * 64 + ks * 32, 64, lrow, hi);
#pragma unroll
            for (int ntl = 0; ntl < 2; ++ntl) {
                v8f acc;
#pragma unroll
                for (int r = 0; r < 8; ++r) acc[r] = 0.f;
#pragma unroll
                for (int ks = 0; ks < 2; ++ks) acc = wmma_bf16(Af[ks], Bf[ntl][ks], acc);
                const int n = n0 + ntl * 16 + lrow;
#pragma unroll
                for (int r = 0; r < 8; ++r)
                    sF1[(mt * 16 + r + 8 * hi) * 256 + n] = (bf16)gelu_exact(acc[r] + bias[ntl]);
            }
        }
    }
    __syncthreads();

    // ================= GEMM2: f2 = gelu(f1 @ W2 + b2)  (K=256) =================
    {
        const int n0 = wv * 32;
#pragma unroll
        for (int ntl = 0; ntl < 2; ++ntl) {
            const int n = n0 + ntl * 16 + lrow;
            v8f acc[4];
#pragma unroll
            for (int mt = 0; mt < 4; ++mt)
#pragma unroll
                for (int r = 0; r < 8; ++r) acc[mt][r] = 0.f;
#pragma unroll
            for (int ks = 0; ks < 8; ++ks) {
                v16bf Bf = ld_frag(W2c + (size_t)(n0 + ntl * 16) * 256 + ks * 32, 256, lrow, hi);
#pragma unroll
                for (int mt = 0; mt < 4; ++mt) {
                    v16bf Af = ld_frag(sF1 + (size_t)(mt * 16) * 256 + ks * 32, 256, lrow, hi);
                    acc[mt] = wmma_bf16(Af, Bf, acc[mt]);
                }
            }
            const float bb = b2g[n];
#pragma unroll
            for (int mt = 0; mt < 4; ++mt)
#pragma unroll
                for (int r = 0; r < 8; ++r)
                    sF2[(mt * 16 + r + 8 * hi) * 256 + n] = (bf16)gelu_exact(acc[mt][r] + bb);
        }
    }
    __syncthreads();

    // ================= GEMM3: v = f2 @ Wv + bv  (N=16, waves 0..3) =================
    if (wv < 4) {
        const int mt = wv;
        v8f acc;
#pragma unroll
        for (int r = 0; r < 8; ++r) acc[r] = 0.f;
#pragma unroll
        for (int ks = 0; ks < 8; ++ks) {
            v16bf Af = ld_frag(sF2 + (size_t)(mt * 16) * 256 + ks * 32, 256, lrow, hi);
            v16bf Bf = ld_frag(Wvc + ks * 32, 256, lrow, hi);
            acc = wmma_bf16(Af, Bf, acc);
        }
        const float bb = bvg[lrow];
#pragma unroll
        for (int r = 0; r < 8; ++r)
            sV[(mt * 16 + r + 8 * hi) * 16 + lrow] = acc[r] + bb;
    }
    __syncthreads();

    // ================= Row phase 1: router softmax, c_bar, v_local =================
    if (tid < 64) {
        const int j = tid;
        const size_t rowG = rowBase + j;
        float rv[16];
#pragma unroll
        for (int d = 0; d < 16; ++d) rv[d] = sV[j * 16 + d];
        float sc[8]; float mx = -3.0e38f; int kch = 0;
#pragma unroll
        for (int c = 0; c < 8; ++c) {
            float s = 0.f;
#pragma unroll
            for (int d = 0; d < 16; ++d) s += rv[d] * ccg[c * 16 + d];
            s *= 0.25f;  // 1/sqrt(16)
            sc[c] = s;
            if (s > mx) { mx = s; kch = c; }
        }
        float sum = 0.f, rw[8];
#pragma unroll
        for (int c = 0; c < 8; ++c) { rw[c] = __expf(sc[c] - mx); sum += rw[c]; }
        const float inv = 1.0f / sum;
#pragma unroll
        for (int c = 0; c < 8; ++c) {
            rw[c] *= inv;
            sRw[j * 8 + c] = rw[c];
            out[OFF_RW + rowG * 8 + c] = rw[c];
        }
        sKch[j] = kch;
        ((int*)out)[OFF_KCHART + rowG] = kch;
#pragma unroll
        for (int d = 0; d < 16; ++d) {
            float cb = 0.f;
#pragma unroll
            for (int c = 0; c < 8; ++c) cb += rw[c] * ccg[c * 16 + d];
            sCb[j * 16 + d] = cb;
            out[OFF_CBAR + rowG * 16 + d] = cb;
            const float vl = rv[d] - cb;
            sVl[j * 16 + d] = vl;
            sVlB[j * 32 + d] = (bf16)vl;
            sVlB[j * 32 + 16 + d] = (bf16)0.0f;  // K padding
        }
    }
    __syncthreads();

    // ====== Dist phase (WMMA): score = code @ v_local^T ; argmin(||c||^2 - 2*score) ======
    {
        const int c = wv;  // one chart per wave
        v16bf Af[4];
#pragma unroll
        for (int mt = 0; mt < 4; ++mt)
            Af[mt] = ld_frag(codeB + (size_t)(c * 64 + mt * 16) * 32, 32, lrow, hi);
#pragma unroll
        for (int nt = 0; nt < 4; ++nt) {
            v16bf Bf = ld_frag(sVlB + (size_t)(nt * 16) * 32, 32, lrow, hi);
            float vmin = 3.0e38f; int imin = 0;
#pragma unroll
            for (int mt = 0; mt < 4; ++mt) {
                v8f acc;
#pragma unroll
                for (int r = 0; r < 8; ++r) acc[r] = 0.f;
                acc = wmma_bf16(Af[mt], Bf, acc);
                const float* cn = codeNrm + c * 64 + mt * 16 + 8 * hi;
#pragma unroll
                for (int r = 0; r < 8; ++r) {
                    const float val = cn[r] - 2.0f * acc[r];   // + const ||v||^2
                    const int kidx = mt * 16 + 8 * hi + r;
                    if (val < vmin) { vmin = val; imin = kidx; }
                }
            }
            // combine the two code halves (lane l <-> l^16 hold same row)
            const float ov = __shfl_xor(vmin, 16, 32);
            const int   oi = __shfl_xor(imin, 16, 32);
            if (ov < vmin || (ov == vmin && oi < imin)) { vmin = ov; imin = oi; }
            if (hi == 0) {
                const int row = nt * 16 + lrow;
                sIdx[c * 64 + row] = imin;
                ((int*)out)[OFF_IDX + (size_t)(rowBase + row) * 8 + c] = imin;
            }
        }
    }
    __syncthreads();

    // ====== Phase 2: per-(row,chart) delta, vq-loss partial, subfield MLP ======
    float lossAcc = 0.f;
#pragma unroll
    for (int rep = 0; rep < 2; ++rep) {
        const int p = tid + rep * 256;          // 512 (row,chart) pairs
        const int row = p >> 3, c = p & 7;
        const float rw = sRw[row * 8 + c];
        const int idx = sIdx[c * 64 + row];
        const float* cd = cbk + (size_t)(c * 64 + idx) * 16;
        float delta[16]; float l2 = 0.f;
#pragma unroll
        for (int d = 0; d < 16; ++d) {
            const float dl = sVl[row * 16 + d] - cd[d];
            delta[d] = dl; l2 += dl * dl;
        }
        lossAcc += rw * l2;
        float h[8];
#pragma unroll
        for (int hh = 0; hh < 8; ++hh) {
            float t = sfb1[hh];
#pragma unroll
            for (int d = 0; d < 16; ++d) t += delta[d] * sfW1[d * 8 + hh];
            h[hh] = gelu_exact(t);
        }
        const size_t ob = OFF_ZNALL_OUT + ((size_t)(rowBase + row) * 8 + c) * 16;
#pragma unroll
        for (int d = 0; d < 16; ++d) {
            float z = sfb2[d];
#pragma unroll
            for (int hh = 0; hh < 8; ++hh) z += h[hh] * sfW2[hh * 16 + d];
            sZnAll[(row * 8 + c) * 16 + d] = z;
            out[ob + d] = z;
        }
    }
    sRed[tid] = lossAcc;
    __syncthreads();

    // ====== Phase 3: blends and final per-row outputs ======
    if (tid < 64) {
        const int j = tid;
        const size_t rowG = rowBase + j;
        float zq[16], zn[16];
#pragma unroll
        for (int d = 0; d < 16; ++d) { zq[d] = 0.f; zn[d] = 0.f; }
#pragma unroll
        for (int c = 0; c < 8; ++c) {
            const float rw = sRw[j * 8 + c];
            const float* cd = cbk + (size_t)(c * 64 + sIdx[c * 64 + j]) * 16;
#pragma unroll
            for (int d = 0; d < 16; ++d) {
                zq[d] += rw * cd[d];
                zn[d] += rw * sZnAll[(j * 8 + c) * 16 + d];
            }
        }
#pragma unroll
        for (int d = 0; d < 16; ++d) {
            const float vl = sVl[j * 16 + d];
            const float cb = sCb[j * 16 + d];
            out[OFF_ZN   + rowG * 16 + d] = zn[d];
            out[OFF_ZTEX + rowG * 16 + d] = vl - zq[d] - zn[d];
            out[OFF_ZGEO + rowG * 16 + d] = cb + zq[d] + zn[d];  // z_q_st == z_q_blended numerically
        }
        ((int*)out)[OFF_KCODE + rowG] = sIdx[sKch[j] * 64 + j];
    }
    if (tid == 0) {
        float t = 0.f;
        for (int i = 0; i < 256; ++i) t += sRed[i];
        // vq_loss = (1 + 0.25) * sum(rw * |delta|^2) / (B * D)  (both loss terms equal numerically)
        lossPart[blockIdx.x] = t * (1.25f / (32768.0f * 16.0f));
    }
}

// ---------------- deterministic final loss reduction ----------------
__global__ void atlas_loss_final(const float* __restrict__ lossPart, float* __restrict__ out) {
    if (blockIdx.x == 0 && threadIdx.x == 0) {
        float s = 0.f;
        for (int i = 0; i < 512; ++i) s += lossPart[i];
        out[OFF_LOSS] = s;
    }
}

extern "C" void kernel_launch(void* const* d_in, const int* in_sizes, int n_in,
                              void* d_out, int out_size, void* d_ws, size_t ws_size,
                              hipStream_t stream) {
    (void)in_sizes; (void)n_in; (void)out_size; (void)ws_size;
    const float* x    = (const float*)d_in[0];
    const float* W1   = (const float*)d_in[1];
    const float* b1   = (const float*)d_in[2];
    const float* W2   = (const float*)d_in[3];
    const float* b2   = (const float*)d_in[4];
    const float* Wv   = (const float*)d_in[5];
    const float* bv   = (const float*)d_in[6];
    const float* cc   = (const float*)d_in[7];
    const float* cbk  = (const float*)d_in[8];
    const float* sfW1 = (const float*)d_in[9];
    const float* sfb1 = (const float*)d_in[10];
    const float* sfW2 = (const float*)d_in[11];
    const float* sfb2 = (const float*)d_in[12];

    char* ws = (char*)d_ws;
    bf16*  W1c      = (bf16*)(ws + WS_W1C);
    bf16*  W2c      = (bf16*)(ws + WS_W2C);
    bf16*  Wvc      = (bf16*)(ws + WS_WVC);
    bf16*  codeB    = (bf16*)(ws + WS_CODEB);
    float* codeNrm  = (float*)(ws + WS_CNRM);
    float* lossPart = (float*)(ws + WS_LOSSP);
    float* out      = (float*)d_out;

    atlas_prep<<<402, 256, 0, stream>>>(W1, W2, Wv, cbk, W1c, W2c, Wvc, codeB, codeNrm);
    atlas_main<<<512, 256, 0, stream>>>(x, b1, b2, bv, cc, cbk, sfW1, sfb1, sfW2, sfb2,
                                        W1c, W2c, Wvc, codeB, codeNrm, out, lossPart);
    atlas_loss_final<<<1, 1, 0, stream>>>(lossPart, out);
}
